// Quantizer_5995774345935
// MI455X (gfx1250) — matrix operand order repaired and checked by reading, not compiled, not run on hardware
//
#include <hip/hip_runtime.h>
#include <hip/hip_bf16.h>

typedef __attribute__((ext_vector_type(16))) __bf16        v16bf;
typedef __attribute__((ext_vector_type(16))) unsigned short v16us;
typedef __attribute__((ext_vector_type(8)))  float          v8f;

union BF16V { v16us u; v16bf b; };

#define N_ROWS   32768
#define IN_DIM   1024
#define E_DIM    64
#define CB_SIZE  1024

#if defined(__has_builtin)
#if __has_builtin(__builtin_amdgcn_tensor_load_to_lds)
#define HAS_TDM 1
#endif
#endif

#ifdef HAS_TDM
typedef __attribute__((ext_vector_type(4))) unsigned int tdm_g0_t;
typedef __attribute__((ext_vector_type(8))) int          tdm_g1_t;
typedef __attribute__((ext_vector_type(4))) int          tdm_g2_t;

// 2D tile (bf16 elements) global -> LDS via Tensor Data Mover, with LDS row padding.
// Descriptor bitfields per CDNA5 ISA 8.3/8.4 (D# groups 0/1); groups 2/3 zero (<=2D).
// This toolchain exposes the 6-arg builtin: (g0 v4u, g1 v8i, g2 v4i, g3 v4i, v8i, i32 cpol).
static __device__ inline void tdm_load_2d_bf16(unsigned lds_addr, const void* gaddr,
                                               unsigned tile_d0, unsigned tile_d1,
                                               unsigned tensor_d0, unsigned tensor_d1,
                                               unsigned stride_d0,
                                               unsigned pad_interval, unsigned pad_amount) {
    unsigned long long ga = (unsigned long long)gaddr;
    tdm_g0_t g0;
    g0[0] = 1u;                                        // count=1, user mode
    g0[1] = lds_addr;                                  // lds_addr [63:32]
    g0[2] = (unsigned)ga;                              // global_addr lo
    g0[3] = (unsigned)(ga >> 32) | 0x80000000u;        // global_addr hi | type=2
    tdm_g1_t g1;
    unsigned f = (1u << 16)                            // data_size = 2 bytes
               | (1u << 20)                            // pad_enable
               | (pad_interval << 22)                  // pad_interval [24:22]
               | (pad_amount << 25);                   // pad_amount  [31:25]
    g1[0] = (int)f;
    g1[1] = (int)((tensor_d0 & 0xFFFFu) << 16);        // tensor_dim0[15:0] @ bits 63:48
    g1[2] = (int)((tensor_d0 >> 16) | ((tensor_d1 & 0xFFFFu) << 16));
    g1[3] = (int)((tensor_d1 >> 16) | (tile_d0 << 16));// tile_dim0 @ bits 127:112
    g1[4] = (int)(tile_d1 & 0xFFFFu);                  // tile_dim1; tile_dim2=0
    g1[5] = (int)stride_d0;                            // tensor_dim0_stride[31:0]
    g1[6] = 0;                                         // stride[47:32] | dim1_stride lo
    g1[7] = 0;
    tdm_g2_t gz4  = {0, 0, 0, 0};
    tdm_g1_t gz8  = {0, 0, 0, 0, 0, 0, 0, 0};
    __builtin_amdgcn_tensor_load_to_lds(g0, g1, gz4, gz4, gz8, 0);
#if __has_builtin(__builtin_amdgcn_s_wait_tensorcnt)
    __builtin_amdgcn_s_wait_tensorcnt(0);
#else
    asm volatile("s_wait_tensorcnt 0x0" ::: "memory");
#endif
}
#endif

static __device__ inline unsigned short f2bf(float f) {
    unsigned u = __float_as_uint(f);
    u += 0x7FFFu + ((u >> 16) & 1u);   // round-to-nearest-even
    return (unsigned short)(u >> 16);
}

static __device__ inline v8f wmma_bf16(const BF16V& A, const BF16V& B, v8f C) {
    return __builtin_amdgcn_wmma_f32_16x16x32_bf16(false, A.b, false, B.b,
                                                   (short)0, C, false, false);
}

// 16 f32 elements of one A-fragment (pre-conversion)
struct AF { float4 x0, x1, x2, x3; };

static __device__ inline AF loadA(const float* ap, int hk) {
    AF r;
    r.x0 = *(const float4*)(ap + hk);
    r.x1 = *(const float4*)(ap + hk + 4);
    r.x2 = *(const float4*)(ap + 16 + hk);
    r.x3 = *(const float4*)(ap + 16 + hk + 4);
    return r;
}

static __device__ inline BF16V cvtA(const AF& a) {
    BF16V A;
    const float* f = (const float*)&a;
#pragma unroll
    for (int j = 0; j < 16; ++j) A.u[j] = f2bf(f[j]);
    return A;
}

// ------- prep: bf16 conversions (codebook, W_post^T, W_pre^T) + loss acc zero -------
__global__ void prep_kernel(const float* __restrict__ cb, const float* __restrict__ wpost,
                            const float* __restrict__ wpre,
                            unsigned short* __restrict__ cbbf,
                            unsigned short* __restrict__ wptbf,
                            unsigned short* __restrict__ wprebf,
                            float* __restrict__ lossAcc) {
    int tid    = blockIdx.x * blockDim.x + threadIdx.x;
    int stride = gridDim.x * blockDim.x;
    if (tid == 0) *lossAcc = 0.0f;
    for (int i = tid; i < CB_SIZE * E_DIM; i += stride)
        cbbf[i] = f2bf(cb[i]);
    // W_post [64,1024] -> transposed bf16 [1024 n][64 k]
    for (int i = tid; i < E_DIM * IN_DIM; i += stride) {
        int k = i >> 10, n = i & (IN_DIM - 1);
        wptbf[n * E_DIM + k] = f2bf(wpost[i]);
    }
    // W_pre [1024,64] -> transposed bf16 [64 n][1024 k]
    for (int i = tid; i < IN_DIM * E_DIM; i += stride) {
        int k = i >> 6, n = i & (E_DIM - 1);
        wprebf[n * IN_DIM + k] = f2bf(wpre[i]);
    }
}

// ---------------- k1: zf = normalize(z @ W_pre + b_pre) ----------------
#define K1_STRIDE 264   // 256 + 8 pad (ushorts); 528B row -> 16B aligned, bank-clean
__global__ __launch_bounds__(256) void k1_pre(const float* __restrict__ z,
                                              const unsigned short* __restrict__ wprebf,
                                              const float* __restrict__ bpre,
                                              float* __restrict__ zf32,
                                              unsigned short* __restrict__ zfbf) {
    __shared__ unsigned short sW[E_DIM * K1_STRIDE];  // [n][k-chunk], bf16
    const int lane  = threadIdx.x & 31;
    const int wave  = threadIdx.x >> 5;
    const int mBase = blockIdx.x * 128 + wave * 16;
    const int m  = lane & 15;
    const int hk = (lane >> 4) << 3;   // A-frag K sub-offset: 0 or 8
    const int n0 = lane & 15;
    const int ko = (lane >> 4) << 3;   // B-frag K sub-offset: 0 or 8

    const float* zrow = z + (size_t)(mBase + m) * IN_DIM;

    v8f acc[4] = {};
    AF af = loadA(zrow, hk);                       // pipeline prologue (kc=0, ks=0)

    for (int kc = 0; kc < IN_DIM; kc += 256) {
        __syncthreads();
        // stage bf16 W_pre^T chunk [64 n][256 k] into padded LDS rows
#ifdef HAS_TDM
        if (threadIdx.x < 32) {
            // row = 256 bf16 = 128 dwords -> pad_interval=6 (2^(6+1)=128); pad 4 dwords -> 3
            tdm_load_2d_bf16((unsigned)(unsigned long long)&sW[0], wprebf + kc,
                             /*tile_d0=*/256, /*tile_d1=*/E_DIM,
                             /*tensor_d0=*/IN_DIM, /*tensor_d1=*/E_DIM,
                             /*stride_d0=*/IN_DIM, /*pad_interval=*/6, /*pad_amount=*/3);
        }
#else
        for (int idx = threadIdx.x; idx < E_DIM * 32; idx += 256) {
            int n = idx >> 5, j = idx & 31;
            *(uint4*)&sW[n * K1_STRIDE + (j << 3)] =
                *(const uint4*)&wprebf[(size_t)n * IN_DIM + kc + (j << 3)];
        }
#endif
        __syncthreads();
        __builtin_prefetch(zrow + kc + 256, 0, 1);  // global_prefetch next z chunk

        for (int ks = 0; ks < 256; ks += 32) {
            // prefetch next A operand (global) before issuing this step's WMMAs
            int kNext = kc + ks + 32;
            AF afn = af;
            if (kNext < IN_DIM) afn = loadA(zrow + kNext, hk);

            BF16V A = cvtA(af);

            // batch all B-fragment LDS loads -> minimal DS waits for 4 WMMAs
            uint4 bq[8];
#pragma unroll
            for (int t = 0; t < 4; ++t) {
                const uint4* bp = (const uint4*)&sW[(t * 16 + n0) * K1_STRIDE + ks + ko];
                bq[2 * t]     = bp[0];
                bq[2 * t + 1] = bp[2];
            }
#pragma unroll
            for (int t = 0; t < 4; ++t) {
                BF16V B;
                ((uint4*)&B.u)[0] = bq[2 * t];
                ((uint4*)&B.u)[1] = bq[2 * t + 1];
                acc[t] = wmma_bf16(A, B, acc[t]);
            }
            af = afn;
        }
    }
    // bias
#pragma unroll
    for (int t = 0; t < 4; ++t) {
        float bb = bpre[t * 16 + n0];
#pragma unroll
        for (int r = 0; r < 8; ++r) acc[t][r] += bb;
    }
    // per-row L2 norm (row = rbase + r; columns spread over 16-lane half-group)
    const int rbase = (lane >> 4) << 3;
    float scale[8];
#pragma unroll
    for (int r = 0; r < 8; ++r) {
        float s = acc[0][r] * acc[0][r] + acc[1][r] * acc[1][r] +
                  acc[2][r] * acc[2][r] + acc[3][r] * acc[3][r];
        s += __shfl_xor(s, 1);
        s += __shfl_xor(s, 2);
        s += __shfl_xor(s, 4);
        s += __shfl_xor(s, 8);
        scale[r] = 1.0f / fmaxf(sqrtf(s), 1e-12f);
    }
#pragma unroll
    for (int r = 0; r < 8; ++r) {
        size_t row = (size_t)(mBase + rbase + r);
#pragma unroll
        for (int t = 0; t < 4; ++t) {
            float v = acc[t][r] * scale[r];
            zf32[row * E_DIM + t * 16 + n0] = v;
            zfbf[row * E_DIM + t * 16 + n0] = f2bf(v);
        }
    }
}

// ---------------- k2: sim = zf @ cbT, argmax -> tokens ----------------
#define K2_STRIDE 72    // 64 + 8 pad (ushorts): 144B rows, 16B aligned, bank-clean
__global__ __launch_bounds__(256) void k2_argmax(const unsigned short* __restrict__ zfbf,
                                                 const unsigned short* __restrict__ cbbf,
                                                 int* __restrict__ tokens) {
    __shared__ unsigned short sC[256 * K2_STRIDE];  // [code][k]
    const int lane  = threadIdx.x & 31;
    const int wave  = threadIdx.x >> 5;
    const int mBase = blockIdx.x * 128 + wave * 16;
    const int m  = lane & 15;
    const int hk = (lane >> 4) << 3;
    const int n0 = lane & 15;
    const int ko = (lane >> 4) << 3;

    // A fragments for K=0..31 and K=32..63 (zf rows, bf16)
    const unsigned short* zp = zfbf + (size_t)(mBase + m) * E_DIM;
    BF16V A0, A1;
    ((uint4*)&A0.u)[0] = *(const uint4*)(zp + hk);
    ((uint4*)&A0.u)[1] = *(const uint4*)(zp + 16 + hk);
    ((uint4*)&A1.u)[0] = *(const uint4*)(zp + 32 + hk);
    ((uint4*)&A1.u)[1] = *(const uint4*)(zp + 48 + hk);

    float mv[8];
    int   mi[8];
#pragma unroll
    for (int r = 0; r < 8; ++r) { mv[r] = -3.4e38f; mi[r] = 0; }

    for (int cc = 0; cc < CB_SIZE; cc += 256) {
        __syncthreads();
#ifdef HAS_TDM
        if (threadIdx.x < 32) {
            // row = 64 bf16 = 32 dwords -> pad_interval=4 (2^(4+1)=32); pad 4 dwords -> 3
            tdm_load_2d_bf16((unsigned)(unsigned long long)&sC[0],
                             cbbf + (size_t)cc * E_DIM,
                             /*tile_d0=*/E_DIM, /*tile_d1=*/256,
                             /*tensor_d0=*/E_DIM, /*tensor_d1=*/CB_SIZE,
                             /*stride_d0=*/E_DIM, /*pad_interval=*/4, /*pad_amount=*/3);
        }
#else
        for (int idx = threadIdx.x; idx < 2048; idx += 256) {
            int c = idx >> 3, j = idx & 7;
            *(uint4*)&sC[c * K2_STRIDE + (j << 3)] =
                *(const uint4*)&cbbf[(size_t)cc * E_DIM + (c << 6) + (j << 3)];
        }
#endif
        __syncthreads();
        for (int t = 0; t < 16; t += 2) {
            // batch LDS loads for two code-tiles
            const uint4* bp0 = (const uint4*)&sC[(t * 16 + n0) * K2_STRIDE + ko];
            const uint4* bp1 = (const uint4*)&sC[((t + 1) * 16 + n0) * K2_STRIDE + ko];
            uint4 q00 = bp0[0], q01 = bp0[2], q02 = bp0[4], q03 = bp0[6];
            uint4 q10 = bp1[0], q11 = bp1[2], q12 = bp1[4], q13 = bp1[6];
            BF16V B00, B01, B10, B11;
            ((uint4*)&B00.u)[0] = q00; ((uint4*)&B00.u)[1] = q01;
            ((uint4*)&B01.u)[0] = q02; ((uint4*)&B01.u)[1] = q03;
            ((uint4*)&B10.u)[0] = q10; ((uint4*)&B10.u)[1] = q11;
            ((uint4*)&B11.u)[0] = q12; ((uint4*)&B11.u)[1] = q13;
            // 4 independent WMMAs (split accumulators)
            v8f a0 = {}, a1 = {}, b0 = {}, b1 = {};
            a0 = wmma_bf16(A0, B00, a0);
            a1 = wmma_bf16(A1, B01, a1);
            b0 = wmma_bf16(A0, B10, b0);
            b1 = wmma_bf16(A1, B11, b1);
            int c0 = cc + t * 16 + n0;
            int c1 = c0 + 16;
#pragma unroll
            for (int r = 0; r < 8; ++r) {
                float v0 = a0[r] + a1[r];
                float v1 = b0[r] + b1[r];
                if (v0 > mv[r]) { mv[r] = v0; mi[r] = c0; }
                if (v1 > mv[r]) { mv[r] = v1; mi[r] = c1; }
            }
        }
    }
    // argmax butterfly across the 16-lane half-group (tie -> lowest index)
#pragma unroll
    for (int r = 0; r < 8; ++r) {
#pragma unroll
        for (int msk = 1; msk < 16; msk <<= 1) {
            float ov = __shfl_xor(mv[r], msk);
            int   oi = __shfl_xor(mi[r], msk);
            if (ov > mv[r] || (ov == mv[r] && oi < mi[r])) { mv[r] = ov; mi[r] = oi; }
        }
    }
    const int rbase = (lane >> 4) << 3;
    if ((lane & 15) == 0) {
#pragma unroll
        for (int r = 0; r < 8; ++r) tokens[mBase + rbase + r] = mi[r];
    }
}

// ---------------- k3: q = codebook[tok] @ W_post + b_post ; loss partials --------
__global__ __launch_bounds__(256) void k3_post(const int* __restrict__ tokens,
                                               const unsigned short* __restrict__ cbbf,
                                               const unsigned short* __restrict__ wptbf,
                                               const float* __restrict__ bpost,
                                               const float* __restrict__ cbf32,
                                               const float* __restrict__ zf32,
                                               float* __restrict__ qout,
                                               float* __restrict__ lossAcc) {
    __shared__ unsigned short sW[256 * K2_STRIDE];  // [n][k]
    const int lane  = threadIdx.x & 31;
    const int wave  = threadIdx.x >> 5;
    const int mBase = blockIdx.x * 128 + wave * 16;
    const int m  = lane & 15;
    const int hk = (lane >> 4) << 3;
    const int n0 = lane & 15;
    const int ko = (lane >> 4) << 3;
    const int rbase = (lane >> 4) << 3;

    // A fragments: gathered quantized rows (bf16 codebook)
    int tok = tokens[mBase + m];
    const unsigned short* ap = cbbf + (size_t)tok * E_DIM;
    BF16V A0, A1;
    ((uint4*)&A0.u)[0] = *(const uint4*)(ap + hk);
    ((uint4*)&A0.u)[1] = *(const uint4*)(ap + 16 + hk);
    ((uint4*)&A1.u)[0] = *(const uint4*)(ap + 32 + hk);
    ((uint4*)&A1.u)[1] = *(const uint4*)(ap + 48 + hk);

    for (int nc = 0; nc < IN_DIM; nc += 256) {
        __syncthreads();
#ifdef HAS_TDM
        if (threadIdx.x < 32) {
            tdm_load_2d_bf16((unsigned)(unsigned long long)&sW[0],
                             wptbf + (size_t)nc * E_DIM,
                             /*tile_d0=*/E_DIM, /*tile_d1=*/256,
                             /*tensor_d0=*/E_DIM, /*tensor_d1=*/IN_DIM,
                             /*stride_d0=*/E_DIM, /*pad_interval=*/4, /*pad_amount=*/3);
        }
#else
        for (int idx = threadIdx.x; idx < 2048; idx += 256) {
            int c = idx >> 3, j = idx & 7;
            *(uint4*)&sW[c * K2_STRIDE + (j << 3)] =
                *(const uint4*)&wptbf[(size_t)nc * E_DIM + (c << 6) + (j << 3)];
        }
#endif
        __syncthreads();
        for (int t = 0; t < 16; t += 2) {
            const uint4* bp0 = (const uint4*)&sW[(t * 16 + n0) * K2_STRIDE + ko];
            const uint4* bp1 = (const uint4*)&sW[((t + 1) * 16 + n0) * K2_STRIDE + ko];
            uint4 q00 = bp0[0], q01 = bp0[2], q02 = bp0[4], q03 = bp0[6];
            uint4 q10 = bp1[0], q11 = bp1[2], q12 = bp1[4], q13 = bp1[6];
            BF16V B00, B01, B10, B11;
            ((uint4*)&B00.u)[0] = q00; ((uint4*)&B00.u)[1] = q01;
            ((uint4*)&B01.u)[0] = q02; ((uint4*)&B01.u)[1] = q03;
            ((uint4*)&B10.u)[0] = q10; ((uint4*)&B10.u)[1] = q11;
            ((uint4*)&B11.u)[0] = q12; ((uint4*)&B11.u)[1] = q13;
            v8f a0 = {}, a1 = {}, b0 = {}, b1 = {};
            a0 = wmma_bf16(A0, B00, a0);
            a1 = wmma_bf16(A1, B01, a1);
            b0 = wmma_bf16(A0, B10, b0);
            b1 = wmma_bf16(A1, B11, b1);
            int col0 = nc + t * 16 + n0;
            int col1 = col0 + 16;
            float bb0 = bpost[col0];
            float bb1 = bpost[col1];
#pragma unroll
            for (int r = 0; r < 8; ++r) {
                size_t row = (size_t)(mBase + rbase + r);
                qout[row * IN_DIM + col0] = a0[r] + a1[r] + bb0;
                qout[row * IN_DIM + col1] = b0[r] + b1[r] + bb1;
            }
        }
    }
    // commitment-loss partial: sum (zf - codebook[tok])^2 over this wave's 16 rows
    int rowL = lane & 15;
    int half = lane >> 4;
    int tok2 = tokens[mBase + rowL];
    const float* zr = zf32  + (size_t)(mBase + rowL) * E_DIM + half * 32;
    const float* cr = cbf32 + (size_t)tok2 * E_DIM + half * 32;
    float s = 0.0f;
#pragma unroll
    for (int j = 0; j < 32; ++j) { float d = zr[j] - cr[j]; s += d * d; }
    s += __shfl_xor(s, 1);
    s += __shfl_xor(s, 2);
    s += __shfl_xor(s, 4);
    s += __shfl_xor(s, 8);
    s += __shfl_xor(s, 16);
    if (lane == 0) atomicAdd(lossAcc, s);
}

// ---------------- k4: finalize loss scalar ----------------
__global__ void k4_finalize(const float* __restrict__ lossAcc, float* __restrict__ outLoss) {
    if (threadIdx.x == 0 && blockIdx.x == 0)
        *outLoss = 0.02f * (*lossAcc) / ((float)N_ROWS * (float)E_DIM);
}

extern "C" void kernel_launch(void* const* d_in, const int* in_sizes, int n_in,
                              void* d_out, int out_size, void* d_ws, size_t ws_size,
                              hipStream_t stream) {
    const float* z     = (const float*)d_in[0];
    const float* cb    = (const float*)d_in[1];
    const float* wpre  = (const float*)d_in[2];
    const float* bpre  = (const float*)d_in[3];
    const float* wpost = (const float*)d_in[4];
    const float* bpost = (const float*)d_in[5];

    float* qout    = (float*)d_out;                                     // [N,1024] f32
    int*   tokens  = (int*)((float*)d_out + (size_t)N_ROWS * IN_DIM);   // [N] i32
    float* lossOut = (float*)d_out + (size_t)N_ROWS * IN_DIM + N_ROWS;  // scalar

    char* w = (char*)d_ws;
    float*          zf32   = (float*)w;          w += (size_t)N_ROWS * E_DIM * 4;
    unsigned short* zfbf   = (unsigned short*)w; w += (size_t)N_ROWS * E_DIM * 2;
    unsigned short* cbbf   = (unsigned short*)w; w += (size_t)CB_SIZE * E_DIM * 2;
    unsigned short* wptbf  = (unsigned short*)w; w += (size_t)IN_DIM * E_DIM * 2;
    unsigned short* wprebf = (unsigned short*)w; w += (size_t)IN_DIM * E_DIM * 2;
    float*          lossAcc = (float*)w;

    prep_kernel<<<64, 256, 0, stream>>>(cb, wpost, wpre, cbbf, wptbf, wprebf, lossAcc);
    k1_pre<<<N_ROWS / 128, 256, 0, stream>>>(z, wprebf, bpre, zf32, zfbf);
    k2_argmax<<<N_ROWS / 128, 256, 0, stream>>>(zfbf, cbbf, tokens);
    k3_post<<<N_ROWS / 128, 256, 0, stream>>>(tokens, cbbf, wptbf, bpost, cb, zf32,
                                              qout, lossAcc);
    k4_finalize<<<1, 32, 0, stream>>>(lossAcc, lossOut);
}
